// Stage1_42322607735575
// MI455X (gfx1250) — matrix operand order repaired and checked by reading, not compiled, not run on hardware
//
#include <hip/hip_runtime.h>

// ---------------------------------------------------------------------------
// Problem constants (match reference)
// ---------------------------------------------------------------------------
#define S_LEN 256
#define T_LEN 256
#define BATCH 32
#define DIN   2048
#define HDIM  1024
#define G4    4096   // 4*H

typedef __attribute__((ext_vector_type(16))) __bf16 v16bf;
typedef __attribute__((ext_vector_type(8)))  __bf16 v8bf;
typedef __attribute__((ext_vector_type(8)))  float  v8f;
typedef __attribute__((ext_vector_type(4)))  int    v4i;

#ifndef __has_builtin
#define __has_builtin(x) 0
#endif

#if __has_builtin(__builtin_amdgcn_global_load_async_to_lds_b128) && \
    __has_builtin(__builtin_amdgcn_s_wait_asynccnt)
#define USE_ASYNC_LDS 1
#else
#define USE_ASYNC_LDS 0
#endif

#if USE_ASYNC_LDS
// GLOBAL_LOAD_ASYNC_TO_LDS_B128: param0 = v4i addrspace(1)*, param1 = v4i
// addrspace(3)*. Generic->AS casts go through uintptr_t (LDS generic address
// keeps the LDS offset in its low 32 bits per the aperture mapping).
__device__ __forceinline__ void async_copy16(const __bf16* g, __bf16* l) {
  auto* gp = reinterpret_cast<__attribute__((address_space(1))) v4i*>(
      reinterpret_cast<uintptr_t>(g));
  auto* lp = reinterpret_cast<__attribute__((address_space(3))) v4i*>(
      static_cast<unsigned int>(reinterpret_cast<uintptr_t>(l)));
  __builtin_amdgcn_global_load_async_to_lds_b128(gp, lp, 0, 0);
}
#endif

// ---------------------------------------------------------------------------
// WMMA fragment helpers (CDNA5 wave32 layouts, CDNA5 ISA 7.12.2)
// ---------------------------------------------------------------------------
// 16-bit A 16x32 (MxK): lane L holds row M=L%16; elements are two contiguous
// 8-element runs: k = kb+off..+7 and kb+off+16..+23, off = (L>=16 ? 8 : 0).
// => two 16-byte vector loads per fragment.
__device__ __forceinline__ v16bf load_a_frag(const __bf16* base, int ld,
                                             int m0, int kbase) {
  const int lane = threadIdx.x & 31;
  const __bf16* p = base + (size_t)(m0 + (lane & 15)) * ld + kbase
                         + ((lane & 16) ? 8 : 0);
  v8bf lo = *(const v8bf*)p;         // k = +0..7
  v8bf hi = *(const v8bf*)(p + 16);  // k = +16..23
  return __builtin_shufflevector(lo, hi, 0, 1, 2, 3, 4, 5, 6, 7,
                                 8, 9, 10, 11, 12, 13, 14, 15);
}

// 16-bit B 32x16 (KxN): lane L holds column N=L%16; same k order as A:
//   k = kb+off..+7 and kb+off+16..+23, off = (L>=16 ? 8 : 0).
__device__ __forceinline__ v16bf load_b_frag(const __bf16* B, int ld,
                                             int n0, int kbase) {
  const int lane = threadIdx.x & 31;
  const __bf16* p = B + (size_t)(n0 + (lane & 15)) * ld + kbase
                      + ((lane & 16) ? 8 : 0);
  v8bf lo = *(const v8bf*)p;
  v8bf hi = *(const v8bf*)(p + 16);
  return __builtin_shufflevector(lo, hi, 0, 1, 2, 3, 4, 5, 6, 7,
                                 8, 9, 10, 11, 12, 13, 14, 15);
}

__device__ __forceinline__ v8f wmma_bf16(v16bf a, v16bf b, v8f c) {
  return __builtin_amdgcn_wmma_f32_16x16x32_bf16(
      /*neg_a=*/false, a, /*neg_b=*/false, b,
      /*c_mod=*/(short)0, c, /*reuse_a=*/false, /*reuse_b=*/false);
}

__device__ __forceinline__ float sigf(float x) {
  return 1.0f / (1.0f + __expf(-x));
}

// ---------------------------------------------------------------------------
// Batched input projection: C[M,N] = A[M,K] * B[N,K]^T + bias[n]
// 64x64 C block per workgroup, 8 waves, 2 tiles/wave.
// A tiles double-buffered in LDS; staged via async-to-LDS when available.
// ---------------------------------------------------------------------------
__device__ __forceinline__ void stage_tile(__bf16* dst, const __bf16* A,
                                           int lda, int m0, int kb, int tid) {
  const int row = tid >> 2, seg = tid & 3;
  const __bf16* src = A + (size_t)(m0 + row) * lda + kb + seg * 16;
  __bf16* d = dst + row * 64 + seg * 16;
#if USE_ASYNC_LDS
  async_copy16(src, d);
  async_copy16(src + 8, d + 8);
#else
  *(uint4*)d       = *(const uint4*)src;
  *(uint4*)(d + 8) = *(const uint4*)(src + 8);
#endif
}

__device__ __forceinline__ void stage_wait_barrier() {
#if USE_ASYNC_LDS
  __builtin_amdgcn_s_wait_asynccnt(0);
#endif
  __syncthreads();
}

__global__ __launch_bounds__(256)
void k_gemm_xw(const __bf16* __restrict__ A, int lda,
               const __bf16* __restrict__ Bm,
               const float* __restrict__ bias,
               float* __restrict__ C, int N, int K) {
  __shared__ __bf16 As[2][64 * 64];   // 2 x 8 KB, double buffered
  const int m0   = blockIdx.y * 64;
  const int n0   = blockIdx.x * 64;
  const int tid  = threadIdx.x;
  const int w    = tid >> 5;
  const int lane = tid & 31;
  const int mt   = w & 3;            // M tile 0..3
  const int ntp  = (w >> 2) * 2;     // N tile pair base: 0 or 2

  v8f acc0 = {};
  v8f acc1 = {};

  stage_tile(As[0], A, lda, m0, 0, tid);
  stage_wait_barrier();

  int buf = 0;
  for (int kb = 0; kb < K; kb += 64, buf ^= 1) {
    if (kb + 64 < K) stage_tile(As[buf ^ 1], A, lda, m0, kb + 64, tid);
#pragma unroll
    for (int kk = 0; kk < 64; kk += 32) {
      v16bf a  = load_a_frag(As[buf], 64, mt * 16, kk);
      v16bf b0 = load_b_frag(Bm, K, n0 + ntp * 16,       kb + kk);
      v16bf b1 = load_b_frag(Bm, K, n0 + (ntp + 1) * 16, kb + kk);
      acc0 = wmma_bf16(a, b0, acc0);
      acc1 = wmma_bf16(a, b1, acc1);
    }
    stage_wait_barrier();
  }

  // Epilogue: C/D f32 layout -> lane holds col n, VGPR r holds row r(+8)
  const int mo = (lane & 16) ? 8 : 0;
  {
    const int n = n0 + ntp * 16 + (lane & 15);
    const float bv = bias[n];
#pragma unroll
    for (int r = 0; r < 8; ++r)
      C[(size_t)(m0 + mt * 16 + r + mo) * N + n] = acc0[r] + bv;
  }
  {
    const int n = n0 + (ntp + 1) * 16 + (lane & 15);
    const float bv = bias[n];
#pragma unroll
    for (int r = 0; r < 8; ++r)
      C[(size_t)(m0 + mt * 16 + r + mo) * N + n] = acc1[r] + bv;
  }
}

// ---------------------------------------------------------------------------
// Fused LSTM step: gates = xW[t] + h@Whh^T ; pointwise c/h update.
// grid = H/16 blocks; 8 waves = {4 gates} x {2 M tiles}; K=1024.
// h double-buffered in bf16 (read hin, write hout); c updated in place.
// ---------------------------------------------------------------------------
__global__ __launch_bounds__(256)
void k_lstm_step(const float* __restrict__ xW,    // [B,4H] for this t
                 const __bf16* __restrict__ Whh,  // [4H,H] bf16
                 const __bf16* __restrict__ hin,  // [B,H]
                 __bf16* __restrict__ hout,       // [B,H]
                 float* __restrict__ c,           // [B,H] in place
                 float* __restrict__ out,         // [B,H] row of output
                 const float* __restrict__ res) { // residual row or null
  __shared__ float gsm[4][BATCH][16];   // 8 KB
  const int n0   = blockIdx.x * 16;
  const int tid  = threadIdx.x;
  const int w    = tid >> 5;
  const int lane = tid & 31;
  const int gate = w >> 1;
  const int mt   = w & 1;
  const int nb   = gate * HDIM + n0;

  v8f acc = {};
#pragma unroll 8
  for (int kk = 0; kk < HDIM; kk += 32) {
    v16bf a = load_a_frag(hin, HDIM, mt * 16, kk);
    v16bf b = load_b_frag(Whh, HDIM, nb, kk);
    acc = wmma_bf16(a, b, acc);
  }

  {
    const int n  = lane & 15;
    const int mo = (lane & 16) ? 8 : 0;
#pragma unroll
    for (int r = 0; r < 8; ++r)
      gsm[gate][mt * 16 + r + mo][n] = acc[r];
  }
  __syncthreads();

#pragma unroll
  for (int e = tid; e < BATCH * 16; e += 256) {
    const int b  = e >> 4;
    const int n  = e & 15;
    const int ng = n0 + n;
    const size_t bi = (size_t)b * HDIM + ng;
    const float iv = gsm[0][b][n] + xW[(size_t)b * G4 + 0 * HDIM + ng];
    const float fv = gsm[1][b][n] + xW[(size_t)b * G4 + 1 * HDIM + ng];
    const float gv = gsm[2][b][n] + xW[(size_t)b * G4 + 2 * HDIM + ng];
    const float ov = gsm[3][b][n] + xW[(size_t)b * G4 + 3 * HDIM + ng];
    const float cn = sigf(fv) * c[bi] + sigf(iv) * tanhf(gv);
    const float hn = sigf(ov) * tanhf(cn);
    c[bi]    = cn;
    hout[bi] = (__bf16)hn;
    out[bi]  = hn + (res ? res[bi] : 0.0f);
  }
}

// ---------------------------------------------------------------------------
// Elementwise helpers
// ---------------------------------------------------------------------------
__global__ __launch_bounds__(256)
void k_f32_to_bf16(const float* __restrict__ x, __bf16* __restrict__ y,
                   size_t n) {
  size_t i = (size_t)blockIdx.x * 256 + threadIdx.x;
  if (i < n) y[i] = (__bf16)x[i];
}

__global__ __launch_bounds__(256)
void k_mask_bf16(const float* __restrict__ x, const float* __restrict__ m,
                 __bf16* __restrict__ y, size_t n) {
  size_t i = (size_t)blockIdx.x * 256 + threadIdx.x;
  if (i < n) y[i] = (__bf16)(x[i] * m[i]);
}

__global__ __launch_bounds__(256)
void k_bias_sum(const float* __restrict__ a, const float* __restrict__ b,
                float* __restrict__ o, int n) {
  int i = blockIdx.x * 256 + threadIdx.x;
  if (i < n) o[i] = a[i] + b[i];
}

__global__ __launch_bounds__(256)
void k_header(float* __restrict__ o) {
  int i = blockIdx.x * 256 + threadIdx.x;
  if (i < BATCH * HDIM) {
    float v = 0.0f;
    if (i == 0) v = (float)(1 + S_LEN + T_LEN);
    else if (i == 1) v = (float)(S_LEN + 1);
    else if (i == 2) v = (float)(S_LEN + 2);
    else if (i == 3) v = (float)(T_LEN + S_LEN + 1);
    o[i] = v;
  }
}

__global__ __launch_bounds__(256)
void k_copy_src_tgt(const float* __restrict__ inp, float* __restrict__ o) {
  size_t i = (size_t)blockIdx.x * 256 + threadIdx.x;
  const size_t n = (size_t)T_LEN * BATCH * HDIM;
  if (i < n) {
    const int t = (int)(i / (BATCH * HDIM));
    const int r = (int)(i % (BATCH * HDIM));
    const int b = r / HDIM;
    const int h = r % HDIM;
    o[(size_t)(1 + S_LEN) * BATCH * HDIM + i] =
        inp[((size_t)(1 + S_LEN + t) * BATCH + b) * DIN + h];
  }
}

// ---------------------------------------------------------------------------
// Host driver
// ---------------------------------------------------------------------------
static inline unsigned blks(size_t n) { return (unsigned)((n + 255) / 256); }

extern "C" void kernel_launch(void* const* d_in, const int* in_sizes, int n_in,
                              void* d_out, int out_size, void* d_ws,
                              size_t ws_size, hipStream_t stream) {
  const float* inp   = (const float*)d_in[0];
  const float* Wih[3] = {(const float*)d_in[1], (const float*)d_in[5],
                         (const float*)d_in[9]};
  const float* Whh[3] = {(const float*)d_in[2], (const float*)d_in[6],
                         (const float*)d_in[10]};
  const float* bih[3] = {(const float*)d_in[3], (const float*)d_in[7],
                         (const float*)d_in[11]};
  const float* bhh[3] = {(const float*)d_in[4], (const float*)d_in[8],
                         (const float*)d_in[12]};
  const float* mask2 = (const float*)d_in[13];
  const float* mask3 = (const float*)d_in[14];
  float* out = (float*)d_out;

  // Workspace layout (256-byte aligned slabs)
  char* w = (char*)d_ws;
  size_t off = 0;
  auto alloc = [&](size_t bytes) {
    void* p = w + off;
    off += (bytes + 255) & ~(size_t)255;
    return p;
  };
  float*  xW     = (float*)alloc((size_t)S_LEN * BATCH * G4 * 4);    // 134 MB
  float*  out2   = (float*)alloc((size_t)S_LEN * BATCH * HDIM * 4);  // 33.5 MB
  float*  out3   = (float*)alloc((size_t)S_LEN * BATCH * HDIM * 4);  // 33.5 MB
  __bf16* xbf    = (__bf16*)alloc((size_t)S_LEN * BATCH * DIN * 2);  // 33.5 MB
  __bf16* WihB   = (__bf16*)alloc((size_t)G4 * DIN * 2);             // 16 MB
  __bf16* WhhB   = (__bf16*)alloc((size_t)G4 * HDIM * 2);            // 8 MB
  float*  bsum   = (float*)alloc((size_t)G4 * 4);
  __bf16* hA     = (__bf16*)alloc((size_t)BATCH * HDIM * 2);
  __bf16* hB     = (__bf16*)alloc((size_t)BATCH * HDIM * 2);
  float*  cSt    = (float*)alloc((size_t)BATCH * HDIM * 4);
  (void)ws_size; (void)n_in; (void)in_sizes; (void)out_size;

  const int Kin[3] = {DIN, HDIM, HDIM};

  for (int layer = 0; layer < 3; ++layer) {
    const int K = Kin[layer];
    // Weight + bias prep (bf16 weights stay L2-resident: 8-16 MB)
    k_f32_to_bf16<<<blks((size_t)G4 * K), 256, 0, stream>>>(
        Wih[layer], WihB, (size_t)G4 * K);
    k_f32_to_bf16<<<blks((size_t)G4 * HDIM), 256, 0, stream>>>(
        Whh[layer], WhhB, (size_t)G4 * HDIM);
    k_bias_sum<<<blks(G4), 256, 0, stream>>>(bih[layer], bhh[layer], bsum, G4);

    // Layer input in bf16
    if (layer == 0) {
      k_f32_to_bf16<<<blks((size_t)S_LEN * BATCH * DIN), 256, 0, stream>>>(
          inp + (size_t)1 * BATCH * DIN, xbf, (size_t)S_LEN * BATCH * DIN);
    } else {
      const float* src  = (layer == 1) ? out2 : out3;
      const float* msk  = (layer == 1) ? mask2 : mask3;
      k_mask_bf16<<<blks((size_t)S_LEN * BATCH * HDIM), 256, 0, stream>>>(
          src, msk, xbf, (size_t)S_LEN * BATCH * HDIM);
    }

    // Batched input projection: xW = x @ Wih^T + (bih+bhh)
    dim3 gg(G4 / 64, (S_LEN * BATCH) / 64);
    k_gemm_xw<<<gg, 256, 0, stream>>>(xbf, K, WihB, bsum, xW, G4, K);

    // Recurrence
    (void)hipMemsetAsync(hA, 0, (size_t)BATCH * HDIM * 2, stream);
    (void)hipMemsetAsync(hB, 0, (size_t)BATCH * HDIM * 2, stream);
    (void)hipMemsetAsync(cSt, 0, (size_t)BATCH * HDIM * 4, stream);

    float* lout = (layer == 0) ? out2
                : (layer == 1) ? out3
                : out + (size_t)1 * BATCH * HDIM;  // out4 rows -> d_out[1..S]
    const float* lres = (layer == 0) ? nullptr : ((layer == 1) ? out2 : out3);

    for (int t = 0; t < S_LEN; ++t) {
      const __bf16* hin = (t & 1) ? hB : hA;
      __bf16* hout      = (t & 1) ? hA : hB;
      const float* rrow = lres ? lres + (size_t)t * BATCH * HDIM : nullptr;
      k_lstm_step<<<HDIM / 16, 256, 0, stream>>>(
          xW + (size_t)t * BATCH * G4, WhhB, hin, hout, cSt,
          lout + (size_t)t * BATCH * HDIM, rrow);
    }
  }

  // Header row + src/tgt passthrough
  k_header<<<blks(BATCH * HDIM), 256, 0, stream>>>(out);
  k_copy_src_tgt<<<blks((size_t)T_LEN * BATCH * HDIM), 256, 0, stream>>>(inp,
                                                                         out);
}
